// Net_21818433863915
// MI455X (gfx1250) — matrix operand order repaired and checked by reading, not compiled, not run on hardware
//
#include <hip/hip_runtime.h>
#include <hip/hip_bf16.h>

// ---------------------------------------------------------------------------
// Problem constants (reference: N=1024, T=4, O=16, CL=64, num_rollout=8)
// ---------------------------------------------------------------------------
#define NN 1024
#define TT 4
#define OO 16
#define CLs 64
#define NROLL 8

typedef __attribute__((ext_vector_type(16))) _Float16 v16h;
typedef __attribute__((ext_vector_type(8)))  _Float16 v8h;
typedef __attribute__((ext_vector_type(4)))  _Float16 v4h;
typedef __attribute__((ext_vector_type(8)))  float    v8f;
typedef __attribute__((ext_vector_type(4)))  float    v4f;

// f16 weight buffer layout (offsets in halves), fragment-major (see prep_kernel)
#define OFF_SC0   0        // 64x64
#define OFF_SC1   4096     // 64x64
#define OFF_R0P   8192     // 64x128
#define OFF_R0Q   16384    // 64x128
#define OFF_R1    24576    // 128x64
#define OFF_R2    32768    // 64x64
#define OFF_A0P   36864    // 64x128
#define OFF_A0Q   45056    // 64x128
#define OFF_A1    53248    // 128x64
#define OFF_AF0   61440    // 64x64
#define OFF_AF1   65536    // 64x64
#define OFF_AF2   69632    // 64x64
#define OFF_O0    73728    // 128x64
#define OFF_O1    81920    // 64x64
#define WH_TOTAL  86016

// ---------------------------------------------------------------------------
// WMMA fragment loaders (layouts per CDNA5 ISA 7.12.2, wave32)
// ---------------------------------------------------------------------------

// A-matrix 16x32 f16 fragment from an f16 row-major LDS matrix.
// lane: M = lane&15, g = lane>>4. half s holds K=16g+s (matches load_b packing)
// => halves a[0..7] = S[m][k0+16g .. +7], a[8..15] = S[m][k0+16g+8 .. +7]
// Two 16-byte ds loads per fragment.
__device__ __forceinline__ v16h load_a(const _Float16* S, int LDA, int k0, int lane) {
  int m = lane & 15, g = lane >> 4;
  const v8h* p0 = (const v8h*)(S + m * LDA + k0 + 16 * g);
  const v8h* p1 = (const v8h*)(S + m * LDA + k0 + 16 * g + 8);
  v8h x0 = *p0, x1 = *p1;
  v16h a;
#pragma unroll
  for (int i = 0; i < 8; ++i) { a[i] = x0[i]; a[8 + i] = x1[i]; }
  return a;
}

// B-matrix 32x16 f16 fragment, pre-swizzled fragment-major in global memory:
// frag(kc,nt) is 32 lanes x 16 contiguous halves (two b128 loads).
__device__ __forceinline__ v16h load_b(const _Float16* Wf, int NT, int kc, int nt, int lane) {
  const v8h* q = (const v8h*)(Wf + (((kc * NT + nt) * 32) + lane) * 16);
  v8h x0 = q[0], x1 = q[1];
  v16h b;
#pragma unroll
  for (int i = 0; i < 8; ++i) { b[i] = x0[i]; b[8 + i] = x1[i]; }
  return b;
}

// One 16x16 output tile:  D = epilogue( A(16xK) @ W(KxNT*16) + bias )
// ACT: 0 none, 1 relu, 2 tanh.  Epilogue order: bias -> act -> residual.
// C/D layout: lane n = nt*16 + (lane&15); VGPR r holds M = r + 8*(lane>>4).
template <int ACT>
__device__ __forceinline__ void gemm_tile(
    const _Float16* __restrict__ A, int LDA, int K,
    const _Float16* __restrict__ Wf, int NT, int nt,
    const float* __restrict__ bias,
    const float* __restrict__ resid, int LDR,
    float* __restrict__ out, int LDO,
    _Float16* __restrict__ out16, int LDO16,
    int lane)
{
  v8f c = {};
  const int KC = K >> 5;
  for (int kc = 0; kc < KC; ++kc) {
    v16h a = load_a(A, LDA, kc * 32, lane);
    v16h b = load_b(Wf, NT, kc, nt, lane);
    c = __builtin_amdgcn_wmma_f32_16x16x32_f16(false, a, false, b, (short)0, c, false, false);
  }
  const int nn = nt * 16 + (lane & 15);
  const int g  = lane >> 4;
  const float bv = bias ? bias[nn] : 0.0f;
#pragma unroll
  for (int r = 0; r < 8; ++r) {
    const int m = r + 8 * g;
    float v = c[r] + bv;
    if (ACT == 1) v = fmaxf(v, 0.0f);
    else if (ACT == 2) v = tanhf(v);
    if (resid) v += resid[m * LDR + nn];
    if (out)   out[m * LDO + nn] = v;
    if (out16) out16[m * LDO16 + nn] = (_Float16)v;
  }
}

// ---------------------------------------------------------------------------
// Weight prep: f32 row-major -> f16 fragment-major.
// element (k,n) of a KxNc weight -> dst[base + ((kc*NT+nt)*32 + nn + 16*(kk>>4))*16 + (kk&15)]
// ---------------------------------------------------------------------------
__global__ void prep_kernel(
    const float* __restrict__ sc0W, const float* __restrict__ sc1W,
    const float* __restrict__ rel0W, const float* __restrict__ rel1W,
    const float* __restrict__ rel2W, const float* __restrict__ att0W,
    const float* __restrict__ att1W, const float* __restrict__ aff0W,
    const float* __restrict__ aff1W, const float* __restrict__ aff2W,
    const float* __restrict__ out0W, const float* __restrict__ out1W,
    _Float16* __restrict__ dst)
{
  int i = blockIdx.x * blockDim.x + threadIdx.x;
  if (i >= WH_TOTAL) return;
  const float* src; int K, Nc, rowoff = 0, base;
  if      (i < OFF_SC1) { src = sc0W; K = 64;  Nc = 64;  base = OFF_SC0; }
  else if (i < OFF_R0P) { src = sc1W; K = 64;  Nc = 64;  base = OFF_SC1; }
  else if (i < OFF_R0Q) { src = rel0W; K = 64; Nc = 128; base = OFF_R0P; }
  else if (i < OFF_R1 ) { src = rel0W; K = 64; Nc = 128; base = OFF_R0Q; rowoff = 64; }
  else if (i < OFF_R2 ) { src = rel1W; K = 128; Nc = 64; base = OFF_R1;  }
  else if (i < OFF_A0P) { src = rel2W; K = 64; Nc = 64;  base = OFF_R2;  }
  else if (i < OFF_A0Q) { src = att0W; K = 64; Nc = 128; base = OFF_A0P; }
  else if (i < OFF_A1 ) { src = att0W; K = 64; Nc = 128; base = OFF_A0Q; rowoff = 64; }
  else if (i < OFF_AF0) { src = att1W; K = 128; Nc = 64; base = OFF_A1;  }
  else if (i < OFF_AF1) { src = aff0W; K = 64; Nc = 64;  base = OFF_AF0; }
  else if (i < OFF_AF2) { src = aff1W; K = 64; Nc = 64;  base = OFF_AF1; }
  else if (i < OFF_O0 ) { src = aff2W; K = 64; Nc = 64;  base = OFF_AF2; }
  else if (i < OFF_O1 ) { src = out0W; K = 128; Nc = 64; base = OFF_O0;  }
  else                  { src = out1W; K = 64; Nc = 64;  base = OFF_O1;  }
  int local = i - base;
  int k = local / Nc, n2 = local % Nc;
  int NT = Nc >> 4;
  int kc = k >> 5, kk = k & 31, nt = n2 >> 4, nn = n2 & 15;
  float v = src[(k + rowoff) * Nc + n2];
  dst[base + (((kc * NT + nt) * 32) + nn + 16 * (kk >> 4)) * 16 + (kk & 15)] = (_Float16)v;
}

// ---------------------------------------------------------------------------
// Encode: cur = state_codes[:, T-1]  with [..., :4] = x[:, T-1]
// ---------------------------------------------------------------------------
__global__ void encode_kernel(const float* __restrict__ x,
                              const float* __restrict__ encW,
                              const float* __restrict__ encb,
                              float* __restrict__ S0)
{
  int idx = blockIdx.x * blockDim.x + threadIdx.x;
  if (idx >= NN * OO * CLs) return;
  int c = idx & 63, o = (idx >> 6) & 15, n = idx >> 10;
  const float* xv = x + ((n * TT + (TT - 1)) * OO + o) * 4;
  float v;
  if (c < 4) v = xv[c];
  else {
    v = encb[c];
#pragma unroll
    for (int k = 0; k < 4; ++k) v += xv[k] * encW[k * CLs + c];
  }
  S0[idx] = v;
}

// present_states == x (state_codes[..., :4] was overwritten with x)
__global__ void present_kernel(const float* __restrict__ x, float* __restrict__ outp)
{
  int idx = blockIdx.x * blockDim.x + threadIdx.x;
  if (idx < NN * TT * OO * 4) outp[idx] = x[idx];
}

// ---------------------------------------------------------------------------
// Fused core step: one block per sample n, 256 threads = 8 wave32s.
// ---------------------------------------------------------------------------
__global__ __launch_bounds__(256) void core_kernel(
    const float* __restrict__ Sin, float* __restrict__ Sout,
    float* __restrict__ outroll, int rstep,
    const _Float16* __restrict__ wh,
    const float* __restrict__ sc0b, const float* __restrict__ sc1b,
    const float* __restrict__ rel0b, const float* __restrict__ rel1b,
    const float* __restrict__ rel2b,
    const float* __restrict__ att0b, const float* __restrict__ att1b,
    const float* __restrict__ att2W, const float* __restrict__ att2b,
    const float* __restrict__ aff0b, const float* __restrict__ aff1b,
    const float* __restrict__ aff2b,
    const float* __restrict__ out0b, const float* __restrict__ out1b,
    const float* __restrict__ rel0last, const float* __restrict__ att0last)
{
  const int n    = blockIdx.x;
  const int tid  = threadIdx.x;
  // Force the wave id into an SGPR: all tile-dispatch branches become scalar
  // (s_cmp/s_cbranch) instead of exec-mask juggling, and WMMA provably runs
  // with EXEC == all-ones.
  const int wave = __builtin_amdgcn_readfirstlane(tid >> 5);
  const int lane = tid & 31;

  const _Float16* sc0W16 = wh + OFF_SC0;
  const _Float16* sc1W16 = wh + OFF_SC1;
  const _Float16* r0P = wh + OFF_R0P;
  const _Float16* r0Q = wh + OFF_R0Q;
  const _Float16* r1W = wh + OFF_R1;
  const _Float16* r2W = wh + OFF_R2;
  const _Float16* a0P = wh + OFF_A0P;
  const _Float16* a0Q = wh + OFF_A0Q;
  const _Float16* a1W = wh + OFF_A1;
  const _Float16* af0 = wh + OFF_AF0;
  const _Float16* af1 = wh + OFF_AF1;
  const _Float16* af2 = wh + OFF_AF2;
  const _Float16* o0W = wh + OFF_O0;
  const _Float16* o1W = wh + OFF_O1;

  __shared__ __align__(16) float sS[16][64];        // state s (f32)
  __shared__ __align__(16) _Float16 sS16[16][64];   // state s (f16 A-input)
  __shared__ __align__(16) float sH1[16][64];       // h1 (f32, residual)
  __shared__ __align__(16) _Float16 sH116[16][64];  // h1 f16, later dyn f16
  __shared__ __align__(16) float sSD[16][64];       // self_dyn, later f1 (f32)
  __shared__ __align__(16) float sP[16][128];       // P (rel)
  __shared__ __align__(16) float sQ[16][128];       // Q (rel), later u1 f32
  __shared__ __align__(16) float sPa[16][128];      // Pa (att), later res
  __shared__ __align__(16) float sQa[16][128];      // Qa (att)
  __shared__ __align__(16) float sDist[16][16];
  __shared__ __align__(16) float sRelDyn[16][64];   // rel_dyn
  __shared__ __align__(16) _Float16 sW1[8][16][128]; // per-wave r1/a1 (f16); reused post-pairwise
  __shared__ __align__(16) _Float16 sW2[8][16][64];  // per-wave r2/a2 (f16)
  __shared__ __align__(16) float sAtt[8][16];        // per-wave masked attention

  float* nullf = (float*)nullptr;
  _Float16* nullh = (_Float16*)nullptr;

  // ---- prefetch f16 weight fragments into cache (global_prefetch_b8) ----
  for (int off = tid * 64; off < WH_TOTAL; off += 256 * 64)
    __builtin_prefetch(wh + off, 0, 3);

  // ---- load state (f32 + f16 shadow), 4 contiguous elems per thread ----
  {
    int o = tid >> 4, c4 = (tid & 15) * 4;
    v4f sv = *(const v4f*)&Sin[n * 1024 + tid * 4];
    *(v4f*)&sS[o][c4] = sv;
    v4h sh;
#pragma unroll
    for (int cc = 0; cc < 4; ++cc) sh[cc] = (_Float16)sv[cc];
    *(v4h*)&sS16[o][c4] = sh;
  }
  __syncthreads();

  // ---- dist (all 256 pairs) + round-1 WMMA jobs: h1(4), P(8), Q(8), Pa(8), Qa(8)
  {
    int i2 = tid >> 4, j2 = tid & 15;
    float dx = sS[i2][0] - sS[j2][0];
    float dy = sS[i2][1] - sS[j2][1];
    sDist[i2][j2] = dx * dx + dy * dy;
  }
  for (int job = wave; job < 36; job += 8) {
    if (job < 4)
      gemm_tile<1>(&sS16[0][0], 64, 64, sc0W16, 4, job, sc0b, nullf, 0, &sH1[0][0], 64, &sH116[0][0], 64, lane);
    else if (job < 12)
      gemm_tile<0>(&sS16[0][0], 64, 64, r0P, 8, job - 4, rel0b, nullf, 0, &sP[0][0], 128, nullh, 0, lane);
    else if (job < 20)
      gemm_tile<0>(&sS16[0][0], 64, 64, r0Q, 8, job - 12, nullf, nullf, 0, &sQ[0][0], 128, nullh, 0, lane);
    else if (job < 28)
      gemm_tile<0>(&sS16[0][0], 64, 64, a0P, 8, job - 20, att0b, nullf, 0, &sPa[0][0], 128, nullh, 0, lane);
    else
      gemm_tile<0>(&sS16[0][0], 64, 64, a0Q, 8, job - 28, nullf, nullf, 0, &sQa[0][0], 128, nullh, 0, lane);
  }
  __syncthreads();

  // ---- self_dyn = h1 @ sc1W + sc1b + h1 (waves 0-3, overlapped with pairwise)
  if (wave < 4)
    gemm_tile<0>(&sH116[0][0], 64, 64, sc1W16, 4, wave, sc1b, &sH1[0][0], 64, &sSD[0][0], 64, nullh, 0, lane);

  // ---- pairwise relational stage: each wave owns row-chunks i (16 pairs each)
  {
    _Float16* W1 = &sW1[wave][0][0];
    _Float16* W2 = &sW2[wave][0][0];
    float*    Aw = &sAtt[wave][0];
    const int c0 = lane * 4;                  // each lane owns 4 contiguous channels
    for (int i = wave; i < 16; i += 8) {
      // --- attention chain: a1 = relu(Pa[i] + Qa[j] + dist*w + b) ---
      {
        v4f pa = *(const v4f*)&sPa[i][c0];
        v4f al = *(const v4f*)&att0last[c0];
#pragma unroll
        for (int j = 0; j < 16; ++j) {
          v4f qa = *(const v4f*)&sQa[j][c0];
          float d = sDist[i][j];
          v4h h;
#pragma unroll
          for (int cc = 0; cc < 4; ++cc)
            h[cc] = (_Float16)fmaxf(pa[cc] + qa[cc] + d * al[cc], 0.0f);
          *(v4h*)&W1[j * 128 + c0] = h;
        }
      }
      for (int t = 0; t < 4; ++t)
        gemm_tile<1>(W1, 128, 128, a1W, 4, t, att1b, nullf, 0, nullf, 0, W2, 64, lane);
      // att[j] = exp(a2[j] . att2W + b); mask diagonal. 32-lane split dot.
      {
        int j = lane & 15, g = lane >> 4;
        float acc = 0.0f;
#pragma unroll 8
        for (int c2 = 0; c2 < 32; ++c2)
          acc += (float)W2[j * 64 + g * 32 + c2] * att2W[g * 32 + c2];
        acc += __shfl_xor(acc, 16, 32);
        if (lane < 16) Aw[j] = (j == i) ? 0.0f : expf(acc + att2b[0]);
      }
      // --- relation chain: r1 = relu(P[i] + Q[j] + dist*w + b) ---
      {
        v4f pp = *(const v4f*)&sP[i][c0];
        v4f rl = *(const v4f*)&rel0last[c0];
#pragma unroll
        for (int j = 0; j < 16; ++j) {
          v4f qq = *(const v4f*)&sQ[j][c0];
          float d = sDist[i][j];
          v4h h;
#pragma unroll
          for (int cc = 0; cc < 4; ++cc)
            h[cc] = (_Float16)fmaxf(pp[cc] + qq[cc] + d * rl[cc], 0.0f);
          *(v4h*)&W1[j * 128 + c0] = h;
        }
      }
      for (int t = 0; t < 4; ++t)
        gemm_tile<1>(W1, 128, 128, r1W, 4, t, rel1b, nullf, 0, nullf, 0, W2, 64, lane);
      // rel = r2 @ rel2W + rel2b + r2, fused with masked weighted reduction over j
      for (int t = 0; t < 4; ++t) {
        v8f c = {};
        for (int kc = 0; kc < 2; ++kc) {
          v16h a = load_a(W2, 64, kc * 32, lane);
          v16h b = load_b(r2W, 4, kc, t, lane);
          c = __builtin_amdgcn_wmma_f32_16x16x32_f16(false, a, false, b, (short)0, c, false, false);
        }
        int nn = t * 16 + (lane & 15), g = lane >> 4;
        float part = 0.0f;
#pragma unroll
        for (int r = 0; r < 8; ++r) {
          int j = r + 8 * g;
          float relv = c[r] + rel2b[nn] + (float)W2[j * 64 + nn];
          part += relv * Aw[j];
        }
        part += __shfl_xor(part, 16, 32);            // combine the two j-halves
        if (lane < 16) sRelDyn[i][nn] = part;
      }
    }
  }
  __syncthreads();

  // ---- dyn = self_dyn + rel_dyn (f16, into sH116); build [f3|s] right half f16
  {
    _Float16* affs16 = &sW1[2][0][0];                // 16x128 f16
    int o = tid >> 4, c4 = (tid & 15) * 4;
    v4h dv;
#pragma unroll
    for (int cc = 0; cc < 4; ++cc)
      dv[cc] = (_Float16)(sSD[o][c4 + cc] + sRelDyn[o][c4 + cc]);
    *(v4h*)&sH116[o][c4] = dv;
    *(v4h*)&affs16[o * 128 + 64 + c4] = *(const v4h*)&sS16[o][c4];
  }
  __syncthreads();

  {
    _Float16* f1_16  = &sW1[0][0][0];                // 16x64
    _Float16* f2_16  = &sW1[1][0][0];                // 16x64
    _Float16* affs16 = &sW1[2][0][0];                // 16x128
    _Float16* u1_16  = &sW1[3][0][0];                // 16x64

    // ---- f1 = tanh(dyn @ aff0W + b)
    if (wave < 4)
      gemm_tile<2>(&sH116[0][0], 64, 64, af0, 4, wave, aff0b, nullf, 0, &sSD[0][0], 64, f1_16, 64, lane);
    __syncthreads();
    // ---- f2 = tanh(f1 @ aff1W + b) + f1
    if (wave < 4)
      gemm_tile<2>(f1_16, 64, 64, af1, 4, wave, aff1b, &sSD[0][0], 64, nullf, 0, f2_16, 64, lane);
    __syncthreads();
    // ---- f3 = f2 @ aff2W + b   (left half of [f3|s])
    if (wave < 4)
      gemm_tile<0>(f2_16, 64, 64, af2, 4, wave, aff2b, nullf, 0, nullf, 0, affs16, 128, lane);
    __syncthreads();
    // ---- u1 = tanh([f3|s] @ out0W + b)
    if (wave < 4)
      gemm_tile<2>(affs16, 128, 128, o0W, 4, wave, out0b, nullf, 0, &sQ[0][0], 128, u1_16, 64, lane);
    __syncthreads();
    // ---- res = u1 @ out1W + b + u1
    if (wave < 4)
      gemm_tile<0>(u1_16, 64, 64, o1W, 4, wave, out1b, &sQ[0][0], 128, &sPa[0][0], 128, nullh, 0, lane);
    __syncthreads();
  }

  // ---- next state: res with [:, :2] += s[:, :2]; write state + rollout slot
  {
    int o = tid >> 4, c4 = (tid & 15) * 4;
    v4f r = *(const v4f*)&sPa[o][c4];
    if (c4 == 0) { r[0] += sS[o][0]; r[1] += sS[o][1]; }
    *(v4f*)&Sout[n * 1024 + tid * 4] = r;
    if (c4 == 0)
      *(v4f*)&outroll[((n * NROLL + rstep) * OO + o) * 4] = r;
  }
}

// ---------------------------------------------------------------------------
// Host launch
// ---------------------------------------------------------------------------
extern "C" void kernel_launch(void* const* d_in, const int* in_sizes, int n_in,
                              void* d_out, int out_size, void* d_ws, size_t ws_size,
                              hipStream_t stream)
{
  const float* x     = (const float*)d_in[0];
  const float* encW  = (const float*)d_in[1];
  const float* encb  = (const float*)d_in[2];
  const float* sc0W  = (const float*)d_in[3];
  const float* sc0b  = (const float*)d_in[4];
  const float* sc1W  = (const float*)d_in[5];
  const float* sc1b  = (const float*)d_in[6];
  const float* rel0W = (const float*)d_in[7];
  const float* rel0b = (const float*)d_in[8];
  const float* rel1W = (const float*)d_in[9];
  const float* rel1b = (const float*)d_in[10];
  const float* rel2W = (const float*)d_in[11];
  const float* rel2b = (const float*)d_in[12];
  const float* att0W = (const float*)d_in[13];
  const float* att0b = (const float*)d_in[14];
  const float* att1W = (const float*)d_in[15];
  const float* att1b = (const float*)d_in[16];
  const float* att2W = (const float*)d_in[17];
  const float* att2b = (const float*)d_in[18];
  const float* aff0W = (const float*)d_in[19];
  const float* aff0b = (const float*)d_in[20];
  const float* aff1W = (const float*)d_in[21];
  const float* aff1b = (const float*)d_in[22];
  const float* aff2W = (const float*)d_in[23];
  const float* aff2b = (const float*)d_in[24];
  const float* out0W = (const float*)d_in[25];
  const float* out0b = (const float*)d_in[26];
  const float* out1W = (const float*)d_in[27];
  const float* out1b = (const float*)d_in[28];
  (void)in_sizes; (void)n_in; (void)out_size; (void)ws_size;

  char* ws = (char*)d_ws;
  _Float16* wh = (_Float16*)ws;                      // 86016 halves = 172032 B
  float* S0 = (float*)(ws + 172032);                 // NN*16*64 f32 = 4 MB
  float* S1 = S0 + NN * OO * CLs;                    // 4 MB
  float* out = (float*)d_out;

  prep_kernel<<<(WH_TOTAL + 255) / 256, 256, 0, stream>>>(
      sc0W, sc1W, rel0W, rel1W, rel2W, att0W, att1W,
      aff0W, aff1W, aff2W, out0W, out1W, wh);

  encode_kernel<<<(NN * OO * CLs) / 256, 256, 0, stream>>>(x, encW, encb, S0);
  present_kernel<<<(NN * TT * OO * 4) / 256, 256, 0, stream>>>(x, out + NN * NROLL * OO * 4);

  float* Sin = S0;
  float* Sout = S1;
  for (int r = 0; r < NROLL; ++r) {
    core_kernel<<<NN, 256, 0, stream>>>(
        Sin, Sout, out, r, wh,
        sc0b, sc1b, rel0b, rel1b, rel2b,
        att0b, att1b, att2W, att2b,
        aff0b, aff1b, aff2b, out0b, out1b,
        rel0W + 128 * 128, att0W + 128 * 128);
    float* t = Sin; Sin = Sout; Sout = t;
  }
}
